// ScaledDotProductModule_48395691492076
// MI455X (gfx1250) — matrix operand order repaired and checked by reading, not compiled, not run on hardware
//
#include <hip/hip_runtime.h>

// ---------------- problem constants ----------------
constexpr int kB = 16;        // batch
constexpr int kS = 2048;      // sequence
constexpr int kD = 64;        // d_model
constexpr int kH = 1024;      // q/k/v width
constexpr int kQKV = 3 * kH;  // 3072

typedef __attribute__((ext_vector_type(16))) __bf16 v16bf;
typedef __attribute__((ext_vector_type(8)))  float  v8f;
typedef int v2i __attribute__((vector_size(8)));
typedef int v4i __attribute__((vector_size(16)));

__device__ inline unsigned short bf2u(__bf16 h) { union { __bf16 b; unsigned short u; } v; v.b = h; return v.u; }

// ---------- gfx1250 async global->LDS copy (ASYNCcnt-tracked, no VGPR data) ----------
// builtin signature: (v_i addrspace(1)* gsrc, v_i addrspace(3)* lds_dst, imm off, imm cpol)
__device__ inline void async_copy_b128(void* lds_dst, const void* gsrc) {
#if __has_builtin(__builtin_amdgcn_global_load_async_to_lds_b128)
  __builtin_amdgcn_global_load_async_to_lds_b128(
      (__attribute__((address_space(1))) v4i*)gsrc,
      (__attribute__((address_space(3))) v4i*)lds_dst, 0, 0);
#elif __has_builtin(__builtin_amdgcn_global_load_async_to_lds_b64)
  __builtin_amdgcn_global_load_async_to_lds_b64(
      (__attribute__((address_space(1))) v2i*)gsrc,
      (__attribute__((address_space(3))) v2i*)lds_dst, 0, 0);
  __builtin_amdgcn_global_load_async_to_lds_b64(
      (__attribute__((address_space(1))) v2i*)((const char*)gsrc + 8),
      (__attribute__((address_space(3))) v2i*)((char*)lds_dst + 8), 0, 0);
#else
  *(v4i*)lds_dst = *(const v4i*)gsrc;   // synchronous fallback
#endif
}

__device__ inline void async_wait0() {
#if __has_builtin(__builtin_amdgcn_s_wait_asynccnt)
  __builtin_amdgcn_s_wait_asynccnt(0);
#else
  asm volatile("s_wait_asynccnt 0x0" ::: "memory");
#endif
}

// A operand (16x32 bf16) from row-major LDS tile, in this code's contiguous k order (not the ISA table order), shared with the B loaders:
// VGPR r: lanes0-15 K=2r,2r+1 ; lanes16-31 K=2r+16,2r+17 (contiguous k = 16*half + s).
__device__ inline v16bf load_A(const __bf16* base, int ld, int lane) {
  const int m = lane & 15, g = lane >> 4;
  const __bf16* row = base + m * ld;
  v16bf a;
#pragma unroll
  for (int r = 0; r < 8; ++r) {
    const int k0 = 16 * g + 2 * r;
    a[2 * r]     = row[k0];
    a[2 * r + 1] = row[k0 + 1];
  }
  return a;
}

// B operand (32x16 bf16) from a TRANSPOSED-stored tile Bt[n][k] (pairs contiguous).
__device__ inline v16bf load_Bt(const __bf16* base, int ld, int lane) {
  const int n = lane & 15, g = lane >> 4;
  const __bf16* col = base + n * ld + 16 * g;
  v16bf b;
#pragma unroll
  for (int r = 0; r < 8; ++r) {
    b[2 * r]     = col[2 * r];
    b[2 * r + 1] = col[2 * r + 1];
  }
  return b;
}

// B operand (32x16 bf16) from a row-major [K][N] tile (strided 16-bit reads).
__device__ inline v16bf load_B_rows(const __bf16* base, int ld, int lane) {
  const int n = lane & 15, g = lane >> 4;
  const __bf16* p = base + 16 * g * ld + n;
  v16bf b;
#pragma unroll
  for (int r = 0; r < 8; ++r) {
    b[2 * r]     = p[(2 * r) * ld];
    b[2 * r + 1] = p[(2 * r + 1) * ld];
  }
  return b;
}

__device__ inline v8f wmma_bf16(v16bf a, v16bf b, v8f c) {
  return __builtin_amdgcn_wmma_f32_16x16x32_bf16(false, a, false, b, (short)0, c, false, false);
}

// horizontal reductions within each 16-lane half-wave (wave32)
__device__ inline float hmax16(float v) {
#pragma unroll
  for (int m = 1; m < 16; m <<= 1) v = fmaxf(v, __shfl_xor(v, m, 32));
  return v;
}
__device__ inline float hsum16(float v) {
#pragma unroll
  for (int m = 1; m < 16; m <<= 1) v += __shfl_xor(v, m, 32);
  return v;
}

// ================= Kernel 1: fused QKV projection =================
// C[BS,3072] = x[BS,64] @ W[64,3072], output bf16 split into q/k/v.
// (Keeps the VGPR path: needs f32 -> bf16 conversion in flight.)
__global__ __launch_bounds__(256) void k_qkv(const float* __restrict__ X,
                                             const float* __restrict__ W,
                                             unsigned short* __restrict__ qo,
                                             unsigned short* __restrict__ ko,
                                             unsigned short* __restrict__ vo) {
  __shared__ __attribute__((aligned(16))) __bf16 Xt[128][72];  // [m][k], +8 pad
  __shared__ __attribute__((aligned(16))) __bf16 Wt[128][72];  // [n][k] transposed
  const int tid = threadIdx.x, lane = tid & 31, w = tid >> 5;
  const int nb = blockIdx.x, mb = blockIdx.y;
  const size_t m0 = (size_t)mb * 128;
  const int n0 = nb * 128;

  for (int i = tid; i < 128 * 64; i += 256) {       // coalesced f32 reads
    const int m = i >> 6, k = i & 63;
    Xt[m][k] = (__bf16)X[(m0 + m) * kD + k];
  }
  for (int i = tid; i < 64 * 128; i += 256) {       // coalesced, transposed store
    const int k = i >> 7, n = i & 127;
    Wt[n][k] = (__bf16)W[(size_t)k * kQKV + n0 + n];
  }
  __syncthreads();

  v8f acc[8] = {};
#pragma unroll
  for (int ks = 0; ks < 2; ++ks) {                  // K = 64 -> 2 wmma steps
    const v16bf a = load_A(&Xt[w * 16][ks * 32], 72, lane);
#pragma unroll
    for (int nt = 0; nt < 8; ++nt) {
      const v16bf bo = load_Bt(&Wt[nt * 16][ks * 32], 72, lane);
      acc[nt] = wmma_bf16(a, bo, acc[nt]);
    }
  }

  const int cn = lane & 15, g = lane >> 4;
#pragma unroll
  for (int nt = 0; nt < 8; ++nt) {
    const int c = n0 + nt * 16 + cn;                // whole 128-block is in one tensor
    unsigned short* dst; int cl;
    if (c < kH)          { dst = qo; cl = c; }
    else if (c < 2 * kH) { dst = ko; cl = c - kH; }
    else                 { dst = vo; cl = c - 2 * kH; }
#pragma unroll
    for (int r = 0; r < 8; ++r) {
      const size_t mg = m0 + w * 16 + r + 8 * g;    // D layout: row = r + 8*half
      dst[mg * kH + cl] = bf2u((__bf16)acc[nt][r]);
    }
  }
}

// ===== Kernel 2: scores (q@k^T, scale, anti-causal mask) + softmax -> P(bf16) =====
// One block = 16 query rows x full 2048 keys. Full score row held in registers:
// wave w owns key columns  t*128 + w*16 .. +16  for t = 0..15 (16 accum tiles).
__global__ __launch_bounds__(256) void k_scores(const unsigned short* __restrict__ qg,
                                                const unsigned short* __restrict__ kg,
                                                unsigned short* __restrict__ Pg) {
  __shared__ __attribute__((aligned(16))) __bf16 Qt[16][1032];  // 16 q rows x 1024 d
  __shared__ __attribute__((aligned(16))) __bf16 Kt[128][72];   // 128 keys x 64 d chunk
  __shared__ float redA[8][16];
  __shared__ float redB[8][16];
  const int tid = threadIdx.x, lane = tid & 31, w = tid >> 5;
  const int qblk = blockIdx.x, b = blockIdx.y;
  const int q0 = qblk * 16;

  { // async-stage this block's 16 q rows (bf16, 16B granules)
    const unsigned short* src = qg + ((size_t)b * kS + q0) * kH;
    for (int c = tid; c < 2048; c += 256) {         // 2048 x 16B = 16 rows x 2KB
      const int m = c >> 7, j = (c & 127) * 8;
      async_copy_b128(&Qt[m][j], src + (size_t)m * kH + j);
    }
  }

  v8f acc[16] = {};
  const int s_start = qblk >> 3;  // key segments fully below the q block are skipped
#pragma unroll
  for (int s = 0; s < 16; ++s) {
    if (s >= s_start) {           // uniform branch per block -> barriers are safe
      const size_t krow0 = (size_t)b * kS + s * 128;
      for (int dc = 0; dc < 16; ++dc) {             // 1024 d in chunks of 64
        __syncthreads();
        const unsigned short* ksrc = kg + krow0 * kH + dc * 64;
        for (int c = tid; c < 1024; c += 256) {     // 128x64 bf16 chunk, async b128
          const int key = c >> 3, j = (c & 7) * 8;
          async_copy_b128(&Kt[key][j], ksrc + (size_t)key * kH + j);
        }
        async_wait0();
        __syncthreads();
#pragma unroll
        for (int ks2 = 0; ks2 < 2; ++ks2) {
          const v16bf a  = load_A(&Qt[0][dc * 64 + ks2 * 32], 1032, lane);
          // B[kk][n] = K[key n][d] : K tile row-major => pairs contiguous
          const v16bf bo = load_Bt(&Kt[w * 16][ks2 * 32], 72, lane);
          acc[s] = wmma_bf16(a, bo, acc[s]);
        }
      }
    }
  }

  // scale 0.25 (1/sqrt(k.shape[0]==16)), anti-causal mask (keep key >= query)
  const int cn = lane & 15, g = lane >> 4;
  float rmax[8];
#pragma unroll
  for (int r = 0; r < 8; ++r) rmax[r] = -3.0e38f;
#pragma unroll
  for (int t = 0; t < 16; ++t) {
#pragma unroll
    for (int r = 0; r < 8; ++r) {
      float sv = acc[t][r] * 0.25f;
      const int key = t * 128 + w * 16 + cn;
      const int qr = q0 + r + 8 * g;
      sv = (key >= qr) ? sv : -9.0e15f;             // matches reference mask value
      acc[t][r] = sv;
      rmax[r] = fmaxf(rmax[r], sv);
    }
  }
#pragma unroll
  for (int r = 0; r < 8; ++r) rmax[r] = hmax16(rmax[r]);  // across wave's 16 cols/row
  if (cn == 0) {                                    // lanes 0 & 16 publish rows 0-7/8-15
#pragma unroll
    for (int r = 0; r < 8; ++r) redA[w][8 * g + r] = rmax[r];
  }
  __syncthreads();
  float gmax[8];
#pragma unroll
  for (int r = 0; r < 8; ++r) {
    const int m = r + 8 * g;
    float v = redA[0][m];
#pragma unroll
    for (int ww = 1; ww < 8; ++ww) v = fmaxf(v, redA[ww][m]);
    gmax[r] = v;
  }

  float rsum[8] = {0.f, 0.f, 0.f, 0.f, 0.f, 0.f, 0.f, 0.f};
#pragma unroll
  for (int t = 0; t < 16; ++t) {
#pragma unroll
    for (int r = 0; r < 8; ++r) {
      const float e = __expf(acc[t][r] - gmax[r]);  // masked -> exp underflows to 0
      acc[t][r] = e;
      rsum[r] += e;
    }
  }
#pragma unroll
  for (int r = 0; r < 8; ++r) rsum[r] = hsum16(rsum[r]);
  if (cn == 0) {
#pragma unroll
    for (int r = 0; r < 8; ++r) redB[w][8 * g + r] = rsum[r];
  }
  __syncthreads();
  float ginv[8];
#pragma unroll
  for (int r = 0; r < 8; ++r) {
    const int m = r + 8 * g;
    float v = 0.f;
#pragma unroll
    for (int ww = 0; ww < 8; ++ww) v += redB[ww][m];
    ginv[r] = 1.0f / v;                             // diag always present -> v >= tiny
  }

  unsigned short* prow = Pg + ((size_t)b * kS + q0) * kS;
#pragma unroll
  for (int t = 0; t < 16; ++t) {
#pragma unroll
    for (int r = 0; r < 8; ++r) {
      const int col = t * 128 + w * 16 + cn;
      const int m = r + 8 * g;
      prow[(size_t)m * kS + col] = bf2u((__bf16)(acc[t][r] * ginv[r]));
    }
  }
}

// ================= Kernel 3: out = P @ V =================
// Per batch: [2048 x 2048] @ [2048 x 1024], skipping all-zero (masked) K chunks.
__global__ __launch_bounds__(256) void k_pv(const unsigned short* __restrict__ Pg,
                                            const unsigned short* __restrict__ vg,
                                            float* __restrict__ outp) {
  __shared__ __attribute__((aligned(16))) __bf16 Pt[128][40];   // [m][kk], +8 pad
  __shared__ __attribute__((aligned(16))) __bf16 Vc[32][136];   // [kk][n], +8 pad
  const int tid = threadIdx.x, lane = tid & 31, w = tid >> 5;
  const int db = blockIdx.x, mb = blockIdx.y, b = blockIdx.z;
  const int n0 = db * 128;
  const int m0 = mb * 128;

  v8f acc[8] = {};
  for (int kc = mb * 4; kc < 64; ++kc) {            // P[q][k]==0 for k<q: skip chunks
    const int k0 = kc * 32;
    __syncthreads();
    const unsigned short* psrc = Pg + ((size_t)b * kS + m0) * kS + k0;
    if (kc + 1 < 64) __builtin_prefetch(psrc + 32, 0, 1);        // next-chunk hint
    for (int c = tid; c < 512; c += 256) {          // 128x32 bf16 P chunk, async b128
      const int m = c >> 2, j = (c & 3) * 8;
      async_copy_b128(&Pt[m][j], psrc + (size_t)m * kS + j);
    }
    const unsigned short* vsrc = vg + ((size_t)b * kS + k0) * kH + n0;
    for (int c = tid; c < 512; c += 256) {          // 32x128 bf16 V chunk, async b128
      const int kk = c >> 4, j = (c & 15) * 8;
      async_copy_b128(&Vc[kk][j], vsrc + (size_t)kk * kH + j);
    }
    async_wait0();
    __syncthreads();
    const v16bf a = load_A(&Pt[w * 16][0], 40, lane);
#pragma unroll
    for (int nt = 0; nt < 8; ++nt) {
      const v16bf bo = load_B_rows(&Vc[0][nt * 16], 136, lane);
      acc[nt] = wmma_bf16(a, bo, acc[nt]);
    }
  }

  const int cn = lane & 15, g = lane >> 4;
  float* orow = outp + ((size_t)b * kS + m0) * kH + n0;
#pragma unroll
  for (int nt = 0; nt < 8; ++nt) {
#pragma unroll
    for (int r = 0; r < 8; ++r) {
      orow[(size_t)(w * 16 + r + 8 * g) * kH + nt * 16 + cn] = acc[nt][r];
    }
  }
}

// ================= host launch =================
extern "C" void kernel_launch(void* const* d_in, const int* in_sizes, int n_in,
                              void* d_out, int out_size, void* d_ws, size_t ws_size,
                              hipStream_t stream) {
  const float* x = (const float*)d_in[0];
  const float* W = (const float*)d_in[1];
  float* out = (float*)d_out;

  // workspace layout (bf16 stored as ushort):
  //   q,k,v: 3 * B*S*H * 2B = 201,326,592 B ; P: B*S*S * 2B = 134,217,728 B
  unsigned short* ws = (unsigned short*)d_ws;
  const size_t nQKV = (size_t)kB * kS * kH;
  unsigned short* qb = ws;
  unsigned short* kb = qb + nQKV;
  unsigned short* vb = kb + nQKV;
  unsigned short* Pb = vb + nQKV;

  k_qkv  <<<dim3(kQKV / 128, (kB * kS) / 128), 256, 0, stream>>>(x, W, qb, kb, vb);
  k_scores<<<dim3(kS / 16, kB),               256, 0, stream>>>(qb, kb, Pb);
  k_pv   <<<dim3(kH / 128, kS / 128, kB),     256, 0, stream>>>(Pb, vb, out);
}
